// SelfAttention_33595234189398
// MI455X (gfx1250) — compile-verified, matrix operand order checked by reading, not run on hardware
//
#include <hip/hip_runtime.h>

typedef __bf16 bf16;
typedef __attribute__((ext_vector_type(16))) __bf16 v16bf;
typedef __attribute__((ext_vector_type(8)))  float  v8f;
typedef __attribute__((ext_vector_type(4)))  unsigned int u32x4;
typedef __attribute__((ext_vector_type(8)))  int i32x8;
typedef __attribute__((ext_vector_type(4)))  int i32x4;

#define NEGV -1000000000.0f

#ifndef __has_builtin
#define __has_builtin(x) 0
#endif
#if __has_builtin(__builtin_amdgcn_tensor_load_to_lds) && \
    __has_builtin(__builtin_amdgcn_s_wait_tensorcnt)
#define HAVE_TDM 1
#else
#define HAVE_TDM 0
#endif

// Load a 16x32 bf16 WMMA fragment half-pair: elements 0..7 from p0, 8..15 from p1.
// Per CDNA5 ISA 16-bit A/B layout: lane m=l%16, K = (l/16)*8 + e  (e<8)
//                                            K = 16 + (l/16)*8 + (e-8)
__device__ __forceinline__ v16bf frag_ld(const bf16* p0, const bf16* p1) {
  v16bf f;
  __builtin_memcpy(&f, p0, 16);
  __builtin_memcpy(reinterpret_cast<char*>(&f) + 16, p1, 16);
  return f;
}

__device__ __forceinline__ v8f wmma_bf16(v16bf a, v16bf b, v8f c) {
  return __builtin_amdgcn_wmma_f32_16x16x32_bf16(false, a, false, b, (short)0, c,
                                                 false, false);
}

// Low 32 bits of a generic pointer to __shared__ == LDS byte offset
// (aperture translation keeps the low 32 address bits).
__device__ __forceinline__ unsigned lds_off(const void* p) {
  return (unsigned)(uintptr_t)p;
}

#if HAVE_TDM
// Issue a TDM 2-D tile load: tile_d1 rows of tile_d0 bf16 elements, row stride
// stride_elems, from gptr into LDS at lds_byte. D# layout per CDNA5 ISA §8.3/8.4.
__device__ __forceinline__ void tdm_load_2d(unsigned lds_byte, const void* gptr,
                                            unsigned tile_d0, unsigned tile_d1,
                                            unsigned stride_elems) {
  unsigned long long ga = (unsigned long long)(uintptr_t)gptr;
  u32x4 g0;
  g0[0] = 1u;                                   // count=1, user descriptor
  g0[1] = lds_byte;                             // lds_addr [63:32]
  g0[2] = (unsigned)ga;                         // global_addr [95:64]
  g0[3] = (unsigned)(ga >> 32) | (2u << 30);    // global_addr hi | type=2
  i32x8 g1;
  g1[0] = 0x00010000;                           // data_size=1 (2 bytes), mask=0
  g1[1] = (int)((stride_elems & 0xFFFFu) << 16);            // tensor_dim0 lo16
  g1[2] = (int)(((stride_elems >> 16) & 0xFFFFu) |
                (0x4000u << 16));               // tensor_dim0 hi | tensor_dim1=16384 lo
  g1[3] = (int)(tile_d0 << 16);                 // tensor_dim1 hi=0 | tile_dim0
  g1[4] = (int)tile_d1;                         // tile_dim1, tile_dim2=0
  g1[5] = (int)stride_elems;                    // tensor_dim0_stride lo32
  g1[6] = 0;                                    // stride hi, dim1_stride lo
  g1[7] = 0;
  i32x4 z4 = {0, 0, 0, 0};
#if __clang_major__ >= 23
  i32x8 z8 = {0, 0, 0, 0, 0, 0, 0, 0};
  __builtin_amdgcn_tensor_load_to_lds(g0, g1, z4, z4, z8, 0);
#else
  __builtin_amdgcn_tensor_load_to_lds(g0, g1, z4, z4, 0);
#endif
}
#endif

// ---------------- fp32 -> bf16 bulk convert ----------------
__global__ void cvt_f32_bf16(const float* __restrict__ src, bf16* __restrict__ dst,
                             int n) {
  int i = (blockIdx.x * blockDim.x + threadIdx.x) * 8;
  if (i >= n) return;
  float4 f0 = *reinterpret_cast<const float4*>(src + i);
  float4 f1 = *reinterpret_cast<const float4*>(src + i + 4);
  bf16 o[8] = {(bf16)f0.x, (bf16)f0.y, (bf16)f0.z, (bf16)f0.w,
               (bf16)f1.x, (bf16)f1.y, (bf16)f1.z, (bf16)f1.w};
  __builtin_memcpy(dst + i, o, 16);
}

// ---------------- bf16 GEMM: out[M,N] = A[M,K] * B[K,N] ----------------
// block tile 128x128, 8 waves of 32x64 each (2x4 WMMA accumulators), K step 32.
// A tile DMA'd by the Tensor Data Mover; B tile transposed into LDS by VALU.
template <bool OUT_F32>
__global__ __launch_bounds__(256) void gemm_bf16_kernel(
    const bf16* __restrict__ A, const bf16* __restrict__ Bm,
    void* __restrict__ outp, int K, int lda, int ldb, int ldo) {
  __shared__ __align__(16) bf16 As[128][32];  // 8KB, row-major [m][k]
  __shared__ __align__(16) bf16 Bt[128][32];  // 8KB, transposed [n][k]

  const int tid = threadIdx.x;
  const int lane = tid & 31;
  const int w = tid >> 5;
  const int wr = w & 3, wc = w >> 2;
  const int m0 = blockIdx.y * 128, n0 = blockIdx.x * 128;
  const int ln = lane & 15;
  const int kh = (lane >> 4) * 8;

  const int brow = tid >> 3;         // 0..31 (k)
  const int bseg = (tid & 7) * 16;   // 0..112 (n)

  v8f acc[2][4] = {};

  for (int k0 = 0; k0 < K; k0 += 32) {
    __syncthreads();
#if HAVE_TDM
    if (w == 0)
      tdm_load_2d(lds_off(&As[0][0]), &A[(size_t)m0 * lda + k0], 32, 128, lda);
#else
    {
      int arow = tid >> 1, aseg = (tid & 1) * 16;
      *reinterpret_cast<uint4*>(&As[arow][aseg]) = *reinterpret_cast<const uint4*>(
          &A[(size_t)(m0 + arow) * lda + k0 + aseg]);
    }
#endif
    // stage B tile transposed
    bf16 tmp[16];
    __builtin_memcpy(tmp, &Bm[(size_t)(k0 + brow) * ldb + n0 + bseg], 32);
#pragma unroll
    for (int t = 0; t < 16; ++t) Bt[bseg + t][brow] = tmp[t];
#if HAVE_TDM
    if (w == 0) __builtin_amdgcn_s_wait_tensorcnt(0);
#endif
    __syncthreads();

    v16bf af[2], bfv[4];
#pragma unroll
    for (int mi = 0; mi < 2; ++mi) {
      const bf16* r = &As[wr * 32 + mi * 16 + ln][0];
      af[mi] = frag_ld(r + kh, r + 16 + kh);
    }
#pragma unroll
    for (int ni = 0; ni < 4; ++ni) {
      const bf16* r = &Bt[wc * 64 + ni * 16 + ln][0];
      bfv[ni] = frag_ld(r + kh, r + 16 + kh);
    }
#pragma unroll
    for (int mi = 0; mi < 2; ++mi)
#pragma unroll
      for (int ni = 0; ni < 4; ++ni)
        acc[mi][ni] = wmma_bf16(af[mi], bfv[ni], acc[mi][ni]);
  }

  const int mhalf = (lane >> 4) * 8;
#pragma unroll
  for (int mi = 0; mi < 2; ++mi)
#pragma unroll
    for (int ni = 0; ni < 4; ++ni)
#pragma unroll
      for (int r = 0; r < 8; ++r) {
        int m = m0 + wr * 32 + mi * 16 + mhalf + r;
        int n = n0 + wc * 64 + ni * 16 + ln;
        if (OUT_F32)
          ((float*)outp)[(size_t)m * ldo + n] = acc[mi][ni][r];
        else
          ((bf16*)outp)[(size_t)m * ldo + n] = (bf16)acc[mi][ni][r];
      }
}

// ---------------- flash attention ----------------
// grid: (B*H, T/128). block: 256 threads = 8 waves; wave owns 16 query rows.
__global__ __launch_bounds__(256) void attn_kernel(
    const bf16* __restrict__ qkv, const unsigned char* __restrict__ mask,
    bf16* __restrict__ y) {
  __shared__ __align__(16) bf16 Qs[128][64];    // 16KB
  __shared__ __align__(16) bf16 Kt[64][64];     // [key][d]   8KB
  __shared__ __align__(16) bf16 Vt[64][64];     // [d][key]   8KB (transposed)
  __shared__ __align__(16) bf16 Ps[8][16][64];  // per-wave P 16KB

  const int tid = threadIdx.x;
  const int lane = tid & 31;
  const int w = tid >> 5;
  const int ln = lane & 15;
  const int kh = (lane >> 4) * 8;
  const int mhalf = (lane >> 4) * 8;

  const int bh = blockIdx.x;
  const int b = bh >> 4;
  const int h = bh & 15;
  const int q0 = blockIdx.y * 128;

  const size_t rs = 3072;  // qkv row stride
  const bf16* qbase = qkv + (size_t)b * 2048 * rs + (size_t)h * 64;
  const bf16* kbase = qbase + 1024;
  const bf16* vbase = qbase + 2048;

  // stage Q (128x64) via TDM
#if HAVE_TDM
  if (w == 0) {
    tdm_load_2d(lds_off(&Qs[0][0]), qbase + (size_t)q0 * rs, 64, 128, 3072);
    __builtin_amdgcn_s_wait_tensorcnt(0);
  }
#else
  {
    int row = tid >> 1, seg = (tid & 1) * 32;
    const bf16* src = qbase + (size_t)(q0 + row) * rs + seg;
#pragma unroll
    for (int j = 0; j < 4; ++j)
      *reinterpret_cast<uint4*>(&Qs[row][seg + j * 8]) =
          *reinterpret_cast<const uint4*>(src + j * 8);
  }
#endif
  __syncthreads();

  // hoisted per-wave Q A-fragments (K halves 0..31, 32..63)
  v16bf aq[2];
  {
    const bf16* r = &Qs[w * 16 + ln][0];
    aq[0] = frag_ld(r + kh, r + 16 + kh);
    aq[1] = frag_ld(r + 32 + kh, r + 48 + kh);
  }

  // query-row mask for this lane's 8 rows
  float mrow[8];
#pragma unroll
  for (int r = 0; r < 8; ++r)
    mrow[r] = mask[(size_t)b * 2048 + q0 + w * 16 + mhalf + r] ? 1.f : 0.f;

  float mstat[8], lstat[8];
#pragma unroll
  for (int r = 0; r < 8; ++r) {
    mstat[r] = -INFINITY;
    lstat[r] = 0.f;
  }
  v8f o[4] = {};

  for (int j = 0; j < 32; ++j) {
    const int k0 = j * 64;
    __syncthreads();
    // stage K tile [key][d] via TDM (overlaps the VALU V-transpose below)
#if HAVE_TDM
    if (w == 0)
      tdm_load_2d(lds_off(&Kt[0][0]), kbase + (size_t)k0 * rs, 64, 64, 3072);
#else
    {
      int row = tid >> 2, seg = (tid & 3) * 16;
      const bf16* src = kbase + (size_t)(k0 + row) * rs + seg;
      *reinterpret_cast<uint4*>(&Kt[row][seg]) =
          *reinterpret_cast<const uint4*>(src);
      *reinterpret_cast<uint4*>(&Kt[row][seg + 8]) =
          *reinterpret_cast<const uint4*>(src + 8);
    }
#endif
    // stage V transposed [d][key]
    {
      int key = tid >> 2, dseg = (tid & 3) * 16;
      bf16 tmp[16];
      __builtin_memcpy(tmp, vbase + (size_t)(k0 + key) * rs + dseg, 32);
#pragma unroll
      for (int t = 0; t < 16; ++t) Vt[dseg + t][key] = tmp[t];
    }
#if HAVE_TDM
    if (w == 0) __builtin_amdgcn_s_wait_tensorcnt(0);
#endif
    __syncthreads();

    // S(16x64) = Q * K^T
    v8f s[4] = {};
#pragma unroll
    for (int kk = 0; kk < 2; ++kk)
#pragma unroll
      for (int ni = 0; ni < 4; ++ni) {
        const bf16* r = &Kt[ni * 16 + ln][0];
        v16bf bfrag = frag_ld(r + kk * 32 + kh, r + kk * 32 + 16 + kh);
        s[ni] = wmma_bf16(aq[kk], bfrag, s[ni]);
      }

    // scale 1/sqrt(64) + query-row mask (branchless, EXEC stays full)
#pragma unroll
    for (int ni = 0; ni < 4; ++ni)
#pragma unroll
      for (int r = 0; r < 8; ++r)
        s[ni][r] = (mrow[r] > 0.f) ? NEGV : s[ni][r] * 0.125f;

    // online softmax per row (rows live across 16-lane halves)
#pragma unroll
    for (int r = 0; r < 8; ++r) {
      float mc = fmaxf(fmaxf(s[0][r], s[1][r]), fmaxf(s[2][r], s[3][r]));
#pragma unroll
      for (int dd = 1; dd < 16; dd <<= 1) mc = fmaxf(mc, __shfl_xor(mc, dd, 32));
      float mn = fmaxf(mstat[r], mc);
      float alpha = (mstat[r] == -INFINITY) ? 0.f : __expf(mstat[r] - mn);
      float psum = 0.f;
#pragma unroll
      for (int ni = 0; ni < 4; ++ni) {
        float p = __expf(s[ni][r] - mn);
        s[ni][r] = p;
        psum += p;
      }
#pragma unroll
      for (int dd = 1; dd < 16; dd <<= 1) psum += __shfl_xor(psum, dd, 32);
      lstat[r] = lstat[r] * alpha + psum;
      mstat[r] = mn;
#pragma unroll
      for (int ni = 0; ni < 4; ++ni) o[ni][r] *= alpha;
    }

    // P (C-layout) -> per-wave LDS (row-major) for A-operand reshape
#pragma unroll
    for (int ni = 0; ni < 4; ++ni)
#pragma unroll
      for (int r = 0; r < 8; ++r)
        Ps[w][mhalf + r][ni * 16 + ln] = (bf16)s[ni][r];

    // O(16x64) += P(16x64) * V(64x64)
#pragma unroll
    for (int kk = 0; kk < 2; ++kk) {
      const bf16* pr = &Ps[w][ln][0];
      v16bf afrag = frag_ld(pr + kk * 32 + kh, pr + kk * 32 + 16 + kh);
#pragma unroll
      for (int ni = 0; ni < 4; ++ni) {
        const bf16* vr = &Vt[ni * 16 + ln][0];
        v16bf bfrag = frag_ld(vr + kk * 32 + kh, vr + kk * 32 + 16 + kh);
        o[ni] = wmma_bf16(afrag, bfrag, o[ni]);
      }
    }
  }

  // normalize and store y[b, q, h*64+d] (bf16)
  float rinv[8];
#pragma unroll
  for (int r = 0; r < 8; ++r) rinv[r] = 1.f / lstat[r];
#pragma unroll
  for (int ni = 0; ni < 4; ++ni)
#pragma unroll
    for (int r = 0; r < 8; ++r) {
      int q = q0 + w * 16 + mhalf + r;
      int dcol = ni * 16 + ln;
      y[((size_t)b * 2048 + q) * 1024 + h * 64 + dcol] =
          (bf16)(o[ni][r] * rinv[r]);
    }
}

extern "C" void kernel_launch(void* const* d_in, const int* in_sizes, int n_in,
                              void* d_out, int out_size, void* d_ws,
                              size_t ws_size, hipStream_t stream) {
  const float* x = (const float*)d_in[0];
  const unsigned char* mask = (const unsigned char*)d_in[1];
  const float* W_attn = (const float*)d_in[2];
  const float* W_proj = (const float*)d_in[3];

  char* ws = (char*)d_ws;
  bf16* xb   = (bf16*)(ws);             // 8192*1024  bf16 = 16 MB
  bf16* wab  = (bf16*)(ws + 16777216);  // 1024*3072  bf16 =  6 MB
  bf16* wpb  = (bf16*)(ws + 23068672);  // 1024*1024  bf16 =  2 MB
  bf16* qkvb = (bf16*)(ws + 25165824);  // 8192*3072  bf16 = 48 MB
  bf16* yb   = (bf16*)(ws + 75497472);  // 8192*1024  bf16 = 16 MB

  cvt_f32_bf16<<<4096, 256, 0, stream>>>(x, xb, 8388608);
  cvt_f32_bf16<<<1536, 256, 0, stream>>>(W_attn, wab, 3145728);
  cvt_f32_bf16<<<512, 256, 0, stream>>>(W_proj, wpb, 1048576);

  // qkv = x @ W_attn   (M=8192, N=3072, K=1024)
  gemm_bf16_kernel<false><<<dim3(24, 64), 256, 0, stream>>>(
      xb, wab, qkvb, 1024, 1024, 3072, 3072);

  // flash attention -> yb (B*H blocks x 16 query tiles)
  attn_kernel<<<dim3(64, 16), 256, 0, stream>>>(qkvb, mask, yb);

  // out = y @ W_proj   (M=8192, N=1024, K=1024), fp32 output
  gemm_bf16_kernel<true><<<dim3(8, 64), 256, 0, stream>>>(
      yb, wpb, d_out, 1024, 1024, 1024, 1024);
}
